// PilotNet_38560216383511
// MI455X (gfx1250) — 8-bit matrix operand order repaired and checked by reading, compiled, not run on hardware
//
#include <hip/hip_runtime.h>

typedef __attribute__((ext_vector_type(8))) int v8i;

#define W_SCALE (1.0f/7.0f)
#define A_SCALE (6.0f/15.0f)
#define BQ_SCALE (W_SCALE*A_SCALE)

__global__ void quant_wb(const float* __restrict__ w, const float* __restrict__ b,
                         signed char* __restrict__ wq, int* __restrict__ qb,
                         int Cout, int CoutPad, int Cin, int KH, int KW, int KtotPad)
{
    int idx = blockIdx.x * blockDim.x + threadIdx.x;
    int total = CoutPad * KtotPad;
    int Ktot = Cin * KH * KW;
    if (idx < total) {
        int o = idx / KtotPad;
        int k = idx - o * KtotPad;
        int q = 0;
        if (o < Cout && k < Ktot) {
            int ci = k % Cin; int t = k / Cin; int kw = t % KW; int kh = t / KW;
            float v = w[(((size_t)o * Cin + ci) * KH + kh) * KW + kw];
            float r = rintf(v * 7.0f);
            r = fminf(fmaxf(r, -7.0f), 7.0f);
            q = (int)r;
        }
        wq[idx] = (signed char)q;
    }
    if (idx < CoutPad) {
        int qv = 0;
        if (idx < Cout) {
            float r = rintf(b[idx] * 17.5f);
            r = fminf(fmaxf(r, -128.0f), 127.0f);
            qv = (int)r;
        }
        qb[idx] = qv;
    }
}

__global__ __launch_bounds__(256) void conv1_f32(
    const float* __restrict__ x, const signed char* __restrict__ wq,
    const int* __restrict__ qb, unsigned char* __restrict__ out)
{
    const int OW = 106, OH = 79, IW = 320, IH = 240, S = 3;
    int m = blockIdx.x * blockDim.x + threadIdx.x;
    int ow = m % OW; int t = m / OW; int oh = t % OH; int n = t / OH;
    const float* xin = x + ((size_t)n * IH + oh * S) * IW + ow * S;
    float xv[25];
#pragma unroll
    for (int kh = 0; kh < 5; ++kh)
#pragma unroll
        for (int kw = 0; kw < 5; ++kw)
            xv[kh * 5 + kw] = xin[kh * IW + kw];
    unsigned char* o8 = out + (size_t)m * 24;
    for (int c = 0; c < 24; ++c) {
        const signed char* wr = wq + c * 32;
        float acc = 0.0f;
#pragma unroll
        for (int k = 0; k < 25; ++k) acc += xv[k] * (float)wr[k];
        float y = acc * W_SCALE + (float)qb[c] * BQ_SCALE;
        float code = rintf(fmaxf(y, 0.0f) * (1.0f / A_SCALE));
        code = fminf(code, 15.0f);
        o8[c] = (unsigned char)(int)code;
    }
}

template <int CIN, int KH, int KW>
__global__ __launch_bounds__(128) void conv_iu8_wmma(
    const unsigned char* __restrict__ actIn,
    const signed char*  __restrict__ wq,
    const int*          __restrict__ qb,
    unsigned char*      __restrict__ actOut,
    int IH, int IW, int S, int OH, int OW, int Cout, int KtotPad)
{
    constexpr int Ktot = CIN * KH * KW;
    int lane = threadIdx.x & 31;
    int wave = threadIdx.x >> 5;
    int half = lane >> 4;
    int row  = lane & 15;
    int mBase = (blockIdx.x * 4 + wave) * 16;
    int m = mBase + row;
    int ow = m % OW; int t = m / OW; int oh = t % OH; int n = t / OH;
    const unsigned char* inB = actIn + (((size_t)n * IH + oh * S) * IW + ow * S) * CIN;
    int nCol = blockIdx.y * 16 + row;
    const signed char* wB = wq + (size_t)nCol * KtotPad;

    v8i acc = {0, 0, 0, 0, 0, 0, 0, 0};
    for (int kb = 0; kb < Ktot; kb += 64) {
        v8i a, b;
#pragma unroll
        for (int v = 0; v < 8; ++v) {
            int k = kb + (v & 1) * 4 + ((v >> 1) & 1) * 16 + (v >> 2) * 32 + half * 8;
            int d = 0;
            if (k < Ktot) {
                int ci = k % CIN; int r = k / CIN; int kw = r % KW; int kh = r / KW;
                d = *(const int*)(inB + ((size_t)kh * IW + kw) * CIN + ci);
            }
            a[v] = d;
        }
#pragma unroll
        for (int v = 0; v < 8; ++v) {
            int k = kb + (v & 1) * 4 + ((v >> 1) & 1) * 16 + (v >> 2) * 32 + half * 8;
            b[v] = *(const int*)(wB + k);
        }
        acc = __builtin_amdgcn_wmma_i32_16x16x64_iu8(false, a, true, b, acc, false, false);
    }

    if (nCol < Cout) {
        int bias = qb[nCol];
#pragma unroll
        for (int d = 0; d < 8; ++d) {
            int mr = mBase + d + half * 8;
            int t2 = acc[d] + bias;
            int code = 0;
            if (t2 > 0) { code = (2 * t2 + 7) / 14; if (code > 15) code = 15; }
            actOut[(size_t)mr * Cout + nCol] = (unsigned char)code;
        }
    }
}

__global__ __launch_bounds__(128) void gemm_iu8_wmma(
    const unsigned char* __restrict__ A, const signed char* __restrict__ wq,
    const int* __restrict__ qb, unsigned char* __restrict__ out,
    int Kpad, int ldOut)
{
    int lane = threadIdx.x & 31;
    int wave = threadIdx.x >> 5;
    int half = lane >> 4;
    int row  = lane & 15;
    int mBase = (blockIdx.x * 4 + wave) * 16;
    const unsigned char* aR = A + (size_t)(mBase + row) * Kpad;
    int nCol = blockIdx.y * 16 + row;
    const signed char* wB = wq + (size_t)nCol * Kpad;

    v8i acc = {0, 0, 0, 0, 0, 0, 0, 0};
    for (int kb = 0; kb < Kpad; kb += 64) {
        v8i a, b;
#pragma unroll
        for (int v = 0; v < 8; ++v) {
            int k = kb + (v & 1) * 4 + ((v >> 1) & 1) * 16 + (v >> 2) * 32 + half * 8;
            a[v] = *(const int*)(aR + k);
        }
#pragma unroll
        for (int v = 0; v < 8; ++v) {
            int k = kb + (v & 1) * 4 + ((v >> 1) & 1) * 16 + (v >> 2) * 32 + half * 8;
            b[v] = *(const int*)(wB + k);
        }
        acc = __builtin_amdgcn_wmma_i32_16x16x64_iu8(false, a, true, b, acc, false, false);
    }

    int bias = qb[nCol];
#pragma unroll
    for (int d = 0; d < 8; ++d) {
        int mr = mBase + d + half * 8;
        int t2 = acc[d] + bias;
        int code = 0;
        if (t2 > 0) { code = (2 * t2 + 7) / 14; if (code > 15) code = 15; }
        out[(size_t)mr * ldOut + nCol] = (unsigned char)code;
    }
}

__global__ __launch_bounds__(256) void repack_nchw(
    const unsigned char* __restrict__ src, unsigned char* __restrict__ dst)
{
    int idx = blockIdx.x * blockDim.x + threadIdx.x;
    int n = idx / 16640; int r = idx - n * 16640;
    int c = r / 260;     int s = r - c * 260;
    dst[idx] = src[((size_t)n * 260 + s) * 64 + c];
}

__global__ __launch_bounds__(256) void final_linear(
    const unsigned char* __restrict__ A, const signed char* __restrict__ wq,
    const int* __restrict__ qb, float* __restrict__ out)
{
    int n = threadIdx.x;
    const unsigned char* ar = A + (size_t)n * 64;
    int acc = 0;
#pragma unroll
    for (int k = 0; k < 10; ++k) acc += (int)ar[k] * (int)wq[k];
    out[n] = (float)(acc + qb[0]) * BQ_SCALE;
}

extern "C" void kernel_launch(void* const* d_in, const int* in_sizes, int n_in,
                              void* d_out, int out_size, void* d_ws, size_t ws_size,
                              hipStream_t stream)
{
    (void)in_sizes; (void)n_in; (void)out_size; (void)ws_size;
    const float* x   = (const float*)d_in[0];
    const float* w1  = (const float*)d_in[1];  const float* b1  = (const float*)d_in[2];
    const float* w2  = (const float*)d_in[3];  const float* b2  = (const float*)d_in[4];
    const float* w3  = (const float*)d_in[5];  const float* b3  = (const float*)d_in[6];
    const float* w4  = (const float*)d_in[7];  const float* b4  = (const float*)d_in[8];
    const float* w5  = (const float*)d_in[9];  const float* b5  = (const float*)d_in[10];
    const float* fw1 = (const float*)d_in[11]; const float* fb1 = (const float*)d_in[12];
    const float* fw2 = (const float*)d_in[13]; const float* fb2 = (const float*)d_in[14];
    const float* fw3 = (const float*)d_in[15]; const float* fb3 = (const float*)d_in[16];
    const float* owp = (const float*)d_in[17]; const float* obp = (const float*)d_in[18];
    float* out = (float*)d_out;

    char* p = (char*)d_ws;
    auto alloc = [&](size_t bytes) -> char* {
        char* r = p; p += (bytes + 255) & ~(size_t)255; return r;
    };

    unsigned char* act1 = (unsigned char*)alloc((size_t)256 * 79 * 106 * 24);
    unsigned char* act2 = (unsigned char*)alloc((size_t)256 * 38 * 51 * 36);
    unsigned char* act3 = (unsigned char*)alloc((size_t)256 * 17 * 24 * 48);
    unsigned char* act4 = (unsigned char*)alloc((size_t)256 * 15 * 22 * 64);
    unsigned char* act5 = (unsigned char*)alloc((size_t)256 * 13 * 20 * 64);
    unsigned char* a5f  = (unsigned char*)alloc((size_t)256 * 16640);
    unsigned char* f1o  = (unsigned char*)alloc((size_t)256 * 128);
    unsigned char* f2o  = (unsigned char*)alloc((size_t)256 * 64);
    unsigned char* f3o  = (unsigned char*)alloc((size_t)256 * 64);

    signed char* wq1 = (signed char*)alloc(24  * 32);
    signed char* wq2 = (signed char*)alloc(48  * 640);
    signed char* wq3 = (signed char*)alloc(48  * 960);
    signed char* wq4 = (signed char*)alloc(64  * 448);
    signed char* wq5 = (signed char*)alloc(64  * 576);
    signed char* wf1 = (signed char*)alloc((size_t)128 * 16640);
    signed char* wf2 = (signed char*)alloc(64  * 128);
    signed char* wf3 = (signed char*)alloc(64  * 64);
    signed char* wo  = (signed char*)alloc(64);
    int* qb1 = (int*)alloc(24  * 4);  int* qb2 = (int*)alloc(48 * 4);
    int* qb3 = (int*)alloc(48  * 4);  int* qb4 = (int*)alloc(64 * 4);
    int* qb5 = (int*)alloc(64  * 4);  int* qf1 = (int*)alloc(128 * 4);
    int* qf2 = (int*)alloc(64  * 4);  int* qf3 = (int*)alloc(64 * 4);
    int* qbo = (int*)alloc(4);

    auto cdiv = [](int a, int b) { return (a + b - 1) / b; };
    auto quant = [&](const float* w, const float* b, signed char* wq, int* qb,
                     int Cout, int CoutPad, int Cin, int KH, int KW, int KtotPad) {
        int total = CoutPad * KtotPad;
        quant_wb<<<cdiv(total, 256), 256, 0, stream>>>(w, b, wq, qb,
                                                       Cout, CoutPad, Cin, KH, KW, KtotPad);
    };

    quant(w1,  b1,  wq1, qb1,  24,  24,     1, 5, 5,    32);
    quant(w2,  b2,  wq2, qb2,  36,  48,    24, 5, 5,   640);
    quant(w3,  b3,  wq3, qb3,  48,  48,    36, 5, 5,   960);
    quant(w4,  b4,  wq4, qb4,  64,  64,    48, 3, 3,   448);
    quant(w5,  b5,  wq5, qb5,  64,  64,    64, 3, 3,   576);
    quant(fw1, fb1, wf1, qf1, 100, 128, 16640, 1, 1, 16640);
    quant(fw2, fb2, wf2, qf2,  50,  64,   100, 1, 1,   128);
    quant(fw3, fb3, wf3, qf3,  10,  64,    50, 1, 1,    64);
    quant(owp, obp, wo,  qbo,   1,   1,    10, 1, 1,    64);

    conv1_f32<<<8374, 256, 0, stream>>>(x, wq1, qb1, act1);

    conv_iu8_wmma<24, 5, 5><<<dim3(7752, 3), 128, 0, stream>>>(
        act1, wq2, qb2, act2, 79, 106, 2, 38, 51, 36, 640);
    conv_iu8_wmma<36, 5, 5><<<dim3(1632, 3), 128, 0, stream>>>(
        act2, wq3, qb3, act3, 38, 51, 2, 17, 24, 48, 960);
    conv_iu8_wmma<48, 3, 3><<<dim3(1320, 4), 128, 0, stream>>>(
        act3, wq4, qb4, act4, 17, 24, 1, 15, 22, 64, 448);
    conv_iu8_wmma<64, 3, 3><<<dim3(1040, 4), 128, 0, stream>>>(
        act4, wq5, qb5, act5, 15, 22, 1, 13, 20, 64, 576);

    repack_nchw<<<16640, 256, 0, stream>>>(act5, a5f);

    gemm_iu8_wmma<<<dim3(4, 8), 128, 0, stream>>>(a5f, wf1, qf1, f1o, 16640, 128);
    gemm_iu8_wmma<<<dim3(4, 4), 128, 0, stream>>>(f1o, wf2, qf2, f2o,   128,  64);
    gemm_iu8_wmma<<<dim3(4, 4), 128, 0, stream>>>(f2o, wf3, qf3, f3o,    64,  64);

    final_linear<<<1, 256, 0, stream>>>(f3o, wo, qbo, out);
}
